// GatedNNMFTransformerEncoder_88038239634059
// MI455X (gfx1250) — matrix operand order repaired and checked by reading, not compiled, not run on hardware
//
#include <hip/hip_runtime.h>
#include <math.h>
#include <stdint.h>

typedef __bf16 bf16;
typedef __attribute__((ext_vector_type(8)))  __bf16 v8bf;
typedef __attribute__((ext_vector_type(16))) __bf16 v16bf;
typedef __attribute__((ext_vector_type(8)))  float  v8f;

// ---------------------------------------------------------------------------
// gfx1250 async global->LDS copy support (guarded; falls back to sync staging)
// ---------------------------------------------------------------------------
#if __has_builtin(__builtin_amdgcn_global_load_async_to_lds_b128) && \
    __has_builtin(__builtin_amdgcn_s_wait_asynccnt)
#define GFX1250_ASYNC 1
#else
#define GFX1250_ASYNC 0
#endif

#if GFX1250_ASYNC
typedef int v4i_vec __attribute__((vector_size(16)));
typedef __attribute__((address_space(1))) v4i_vec* g_v4i_ptr;   // global int4*
typedef __attribute__((address_space(3))) v4i_vec* l_v4i_ptr;   // LDS int4*
__device__ __forceinline__ void async_cp16(void* lds, const void* g) {
    // AS(1) shares the generic VA; AS(3) pointer value == low 32 bits of the
    // generic VA.
    __builtin_amdgcn_global_load_async_to_lds_b128(
        (g_v4i_ptr)(uintptr_t)g, (l_v4i_ptr)(unsigned)(uintptr_t)lds, 0, 0);
}
template<int N>
__device__ __forceinline__ void wait_async() {
    __builtin_amdgcn_s_wait_asynccnt((unsigned short)N);
}
#endif

__device__ __forceinline__ float gelu_exact(float x) {
    return 0.5f * x * (1.0f + erff(x * 0.70710678118654752440f));
}

// ---------------------------------------------------------------------------
// Batched GEMM: C = A x B (+epilogue), bf16 inputs, f32 accumulate via WMMA.
//  !TA: A is MxK row-major (lda = row stride). TA: A is KxM row-major.
//  !TB: B is KxN row-major.                    TB: B is NxK row-major.
// EPI: 0 = f32 store
//      1 = bf16 store
//      2 = bias + GELU, f32 store
//      3 = bias + GELU, bf16 store
//      4 = bias + residual, f32 store
//      5 = bias + GELU + residual, f32 store
// All M,N,K divide the tile sizes (asserted by launch choices), no bounds checks.
// ---------------------------------------------------------------------------
template<int BM, int BN, int BK, int WR, int WC, bool TA, bool TB, int EPI>
__global__ __launch_bounds__(256) void gemm_bf16_kernel(
    const bf16* __restrict__ A, const bf16* __restrict__ Bm,
    float* __restrict__ Cf, bf16* __restrict__ Cb,
    const float* __restrict__ bias, const float* __restrict__ res,
    int M, int N, int K,
    long lda, long ldb, long ldc,
    long sA, long sB, long sC)
{
    static_assert(BK == 32, "one wmma k-step per tile");
    constexpr int SUBM = (BM / 16) / WR;
    constexpr int SUBN = (BN / 16) / WC;
    constexpr int PAD  = 8;            // row stride 40 bf16 = 80B, 16B-aligned chunks
    constexpr int LDW  = BK + PAD;

#if GFX1250_ASYNC
    constexpr bool PIPE = (!TA) && TB; // the five big GEMMs: double-buffered async
#else
    constexpr bool PIPE = false;
#endif
    constexpr int NBUF = PIPE ? 2 : 1;

    __shared__ bf16 As[NBUF][BM][LDW];
    __shared__ bf16 Bs[NBUF][BN][LDW];

    (void)M; (void)N;
    const bf16* Ab = A  + (long)blockIdx.z * sA;
    const bf16* Bb = Bm + (long)blockIdx.z * sB;
    const int m0 = blockIdx.y * BM;
    const int n0 = blockIdx.x * BN;

    const int tid  = threadIdx.x;
    const int wave = tid >> 5;
    const int lane = tid & 31;
    const int wr = wave % WR;
    const int wc = wave / WR;
    const int lm = lane & 15;
    const int lh = lane >> 4;

    v8f acc[SUBM][SUBN];
    #pragma unroll
    for (int i = 0; i < SUBM; ++i)
        #pragma unroll
        for (int j = 0; j < SUBN; ++j)
            #pragma unroll
            for (int r = 0; r < 8; ++r) acc[i][j][r] = 0.f;

    // ---- fragment loads (ISA VGPR layouts) + WMMA for one staged tile ----
    auto compute_tile = [&](int buf) {
        v16bf afrag[SUBM];
        v16bf bfrag[SUBN];
        #pragma unroll
        for (int i = 0; i < SUBM; ++i) {
            // A 16x32 bf16: M = lane&15; chunks K = lh*8..+7 and lh*8+16..+23
            const bf16* ap = &As[buf][wr * SUBM * 16 + i * 16 + lm][lh * 8];
            v8bf lo = *(const v8bf*)ap;
            v8bf hi = *(const v8bf*)(ap + 16);
            #pragma unroll
            for (int e = 0; e < 8; ++e) { afrag[i][e] = lo[e]; afrag[i][e + 8] = hi[e]; }
        }
        #pragma unroll
        for (int j = 0; j < SUBN; ++j) {
            // B 32x16 bf16: N = lane&15; chunks K = lh*8..+7 and lh*8+16..+23
            const bf16* bp = &Bs[buf][wc * SUBN * 16 + j * 16 + lm][lh * 8];
            v8bf lo = *(const v8bf*)bp;
            v8bf hi = *(const v8bf*)(bp + 16);
            #pragma unroll
            for (int e = 0; e < 8; ++e) { bfrag[j][e] = lo[e]; bfrag[j][e + 8] = hi[e]; }
        }
        #pragma unroll
        for (int i = 0; i < SUBM; ++i)
            #pragma unroll
            for (int j = 0; j < SUBN; ++j)
                acc[i][j] = __builtin_amdgcn_wmma_f32_16x16x32_bf16(
                    false, afrag[i], false, bfrag[j], (short)0, acc[i][j], false, false);
    };

    if constexpr (PIPE) {
#if GFX1250_ASYNC
        constexpr int ITA = (BM * (BK / 8)) / 256;   // 16B chunks per thread (A)
        constexpr int ITB = (BN * (BK / 8)) / 256;   // 16B chunks per thread (B)
        constexpr int PER_TILE = ITA + ITB;          // async instrs outstanding per tile
        static_assert((BM * (BK / 8)) % 256 == 0 && (BN * (BK / 8)) % 256 == 0, "");

        auto stage = [&](int buf, int k0) {
            #pragma unroll
            for (int t = 0; t < ITA; ++t) {
                int c = tid + t * 256;
                int m = c / (BK / 8);
                int kc = (c % (BK / 8)) * 8;
                async_cp16(&As[buf][m][kc], Ab + (long)(m0 + m) * lda + (k0 + kc));
            }
            #pragma unroll
            for (int t = 0; t < ITB; ++t) {
                int c = tid + t * 256;
                int n = c / (BK / 8);
                int kc = (c % (BK / 8)) * 8;
                async_cp16(&Bs[buf][n][kc], Bb + (long)(n0 + n) * ldb + (k0 + kc));
            }
        };

        const int nk = K / BK;
        stage(0, 0);
        for (int ki = 0; ki < nk; ++ki) {
            int buf = ki & 1;
            if (ki + 1 < nk) {
                stage(buf ^ 1, (ki + 1) * BK);   // overlap next tile's DMA
                wait_async<PER_TILE>();          // in-order: tile ki has landed
            } else {
                wait_async<0>();
            }
            __syncthreads();
            compute_tile(buf);
            __syncthreads();                     // all waves done reading buf
        }
#endif
    } else {
        // --------- synchronous staging, compile-time trip counts ---------
        for (int k0 = 0; k0 < K; k0 += BK) {
            if constexpr (!TA) {
                constexpr int IT = (BM * (BK / 8)) / 256;
                v8bf tmp[IT];
                #pragma unroll
                for (int t = 0; t < IT; ++t) {
                    int c = tid + t * 256;
                    int m = c / (BK / 8), kc = (c % (BK / 8)) * 8;
                    tmp[t] = *(const v8bf*)(Ab + (long)(m0 + m) * lda + (k0 + kc));
                }
                #pragma unroll
                for (int t = 0; t < IT; ++t) {
                    int c = tid + t * 256;
                    int m = c / (BK / 8), kc = (c % (BK / 8)) * 8;
                    *(v8bf*)&As[0][m][kc] = tmp[t];
                }
            } else {
                constexpr int EL = (BM * BK) / 256;
                bf16 tmp[EL];
                #pragma unroll
                for (int t = 0; t < EL; ++t) {
                    int e = tid + t * 256;
                    int k = e / BM, m = e % BM;
                    tmp[t] = Ab[(long)(k0 + k) * lda + (m0 + m)];
                }
                #pragma unroll
                for (int t = 0; t < EL; ++t) {
                    int e = tid + t * 256;
                    int k = e / BM, m = e % BM;
                    As[0][m][k] = tmp[t];
                }
            }
            if constexpr (TB) {
                constexpr int IT = (BN * (BK / 8)) / 256;
                v8bf tmp[IT];
                #pragma unroll
                for (int t = 0; t < IT; ++t) {
                    int c = tid + t * 256;
                    int n = c / (BK / 8), kc = (c % (BK / 8)) * 8;
                    tmp[t] = *(const v8bf*)(Bb + (long)(n0 + n) * ldb + (k0 + kc));
                }
                #pragma unroll
                for (int t = 0; t < IT; ++t) {
                    int c = tid + t * 256;
                    int n = c / (BK / 8), kc = (c % (BK / 8)) * 8;
                    *(v8bf*)&Bs[0][n][kc] = tmp[t];
                }
            } else {
                constexpr int EL = (BN * BK) / 256;
                bf16 tmp[EL];
                #pragma unroll
                for (int t = 0; t < EL; ++t) {
                    int e = tid + t * 256;
                    int k = e / BN, n = e % BN;
                    tmp[t] = Bb[(long)(k0 + k) * ldb + (n0 + n)];
                }
                #pragma unroll
                for (int t = 0; t < EL; ++t) {
                    int e = tid + t * 256;
                    int k = e / BN, n = e % BN;
                    Bs[0][n][k] = tmp[t];
                }
            }
            __syncthreads();
            if (k0 + BK < K) { // prefetch next K tile (global_prefetch_b8)
                if constexpr (!TA) { if (tid < BM) __builtin_prefetch(Ab + (long)(m0 + tid) * lda + (k0 + BK), 0, 3); }
                else               { if (tid < BK) __builtin_prefetch(Ab + (long)(k0 + BK + tid) * lda + m0, 0, 3); }
                if constexpr (TB)  { if (tid < BN) __builtin_prefetch(Bb + (long)(n0 + tid) * ldb + (k0 + BK), 0, 3); }
                else               { if (tid < BK) __builtin_prefetch(Bb + (long)(k0 + BK + tid) * ldb + n0, 0, 3); }
            }
            compute_tile(0);
            __syncthreads();
        }
    }

    // ---- epilogue: C layout VGPR r -> M = r + 8*lh, N = lane&15 ----
    const int mBase = m0 + wr * SUBM * 16;
    const int nBase = n0 + wc * SUBN * 16;
    #pragma unroll
    for (int i = 0; i < SUBM; ++i)
        #pragma unroll
        for (int j = 0; j < SUBN; ++j) {
            int n = nBase + j * 16 + lm;
            float bv = (EPI >= 2) ? bias[n] : 0.f;
            #pragma unroll
            for (int r = 0; r < 8; ++r) {
                int m = mBase + i * 16 + lh * 8 + r;
                long idx = (long)blockIdx.z * sC + (long)m * ldc + n;
                float v = acc[i][j][r] + bv;
                if (EPI == 2 || EPI == 3 || EPI == 5) v = gelu_exact(v);
                if (EPI == 4 || EPI == 5) v += res[idx];
                if (EPI == 1 || EPI == 3) Cb[idx] = (bf16)v;
                else                      Cf[idx] = v;
            }
        }
}

// ---------------------------------------------------------------------------
// LayerNorm over a row of width W, output bf16 (optional ReLU). One block/row.
// ---------------------------------------------------------------------------
__global__ __launch_bounds__(256) void ln_bf16_kernel(
    const float* __restrict__ x, long ld,
    const float* __restrict__ g, const float* __restrict__ b,
    bf16* __restrict__ out, long ldo, int W, int relu)
{
    __shared__ float red[256];
    long row = blockIdx.x;
    const float* xr = x + row * ld;
    float s = 0.f, ss = 0.f;
    for (int i = threadIdx.x; i < W; i += 256) { float v = xr[i]; s += v; ss += v * v; }
    red[threadIdx.x] = s; __syncthreads();
    for (int o = 128; o > 0; o >>= 1) { if ((int)threadIdx.x < o) red[threadIdx.x] += red[threadIdx.x + o]; __syncthreads(); }
    float mean = red[0] / W; __syncthreads();
    red[threadIdx.x] = ss; __syncthreads();
    for (int o = 128; o > 0; o >>= 1) { if ((int)threadIdx.x < o) red[threadIdx.x] += red[threadIdx.x + o]; __syncthreads(); }
    float var  = red[0] / W - mean * mean;
    float rstd = rsqrtf(var + 1e-5f);
    bf16* orow = out + row * ldo;
    for (int i = threadIdx.x; i < W; i += 256) {
        float v = (xr[i] - mean) * rstd * g[i] + b[i];
        if (relu) v = fmaxf(v, 0.f);
        orow[i] = (bf16)v;
    }
}

// Row softmax over last dim 64, scaled; one wave per row (8 rows / block).
__global__ __launch_bounds__(256) void softmax64_kernel(
    const float* __restrict__ s, float scale,
    float* __restrict__ cf, bf16* __restrict__ cb, long nrows)
{
    long row = (long)blockIdx.x * 8 + (threadIdx.x >> 5);
    int lane = threadIdx.x & 31;
    if (row >= nrows) return;
    const float* r = s + row * 64;
    float v0 = r[lane] * scale, v1 = r[lane + 32] * scale;
    float m = fmaxf(v0, v1);
    for (int o = 16; o > 0; o >>= 1) m = fmaxf(m, __shfl_xor(m, o, 32));
    float e0 = __expf(v0 - m), e1 = __expf(v1 - m);
    float sum = e0 + e1;
    for (int o = 16; o > 0; o >>= 1) sum += __shfl_xor(sum, o, 32);
    float inv = 1.f / sum;
    e0 *= inv; e1 *= inv;
    cf[row * 64 + lane] = e0;       cf[row * 64 + lane + 32] = e1;
    cb[row * 64 + lane] = (bf16)e0; cb[row * 64 + lane + 32] = (bf16)e1;
}

// t *= num / (den + eps); keep f32 master + bf16 mirror.
__global__ void mul_update_kernel(float* __restrict__ t, bf16* __restrict__ tb,
                                  const float* __restrict__ num, const float* __restrict__ den, long n)
{
    long i = (long)blockIdx.x * 256 + threadIdx.x;
    if (i < n) { float v = t[i] * num[i] / (den[i] + 1e-6f); t[i] = v; tb[i] = (bf16)v; }
}

// g = bf16(z1 * z2r)  (z1 strided inside u, z2r contiguous bf16)
__global__ void gate_mul_kernel(const float* __restrict__ z1, long ld1,
                                const bf16* __restrict__ z2, bf16* __restrict__ g,
                                long n, int W)
{
    long i = (long)blockIdx.x * 256 + threadIdx.x;
    if (i < n) {
        long row = i / W; int col = (int)(i % W);
        g[i] = (bf16)(z1[row * ld1 + col] * (float)z2[i]);
    }
}

__global__ void cvt_bf16_kernel(const float* __restrict__ in, bf16* __restrict__ o, long n)
{
    long i = (long)blockIdx.x * 256 + threadIdx.x;
    if (i < n) o[i] = (bf16)in[i];
}

// broadcast bases0 (per) across batch -> f32 + bf16
__global__ void bases_init_kernel(const float* __restrict__ b0, float* __restrict__ bf_,
                                  bf16* __restrict__ bb, long per, long n)
{
    long i = (long)blockIdx.x * 256 + threadIdx.x;
    if (i < n) { float v = b0[i % per]; bf_[i] = v; bb[i] = (bf16)v; }
}

// ---------------------------------------------------------------------------
// Host-side launch helpers
// ---------------------------------------------------------------------------
template<bool TA, bool TB, int EPI>
static void gemm_big(hipStream_t st, const bf16* A, const bf16* B, float* Cf, bf16* Cb,
                     const float* bias, const float* res, int M, int N, int K,
                     long lda, long ldb, long ldc, long sA, long sB, long sC, int batch)
{
    dim3 g((unsigned)(N / 128), (unsigned)(M / 128), (unsigned)batch);
    gemm_bf16_kernel<128,128,32,4,2,TA,TB,EPI><<<g, dim3(256), 0, st>>>(
        A, B, Cf, Cb, bias, res, M, N, K, lda, ldb, ldc, sA, sB, sC);
}

template<bool TA, bool TB, int EPI>
static void gemm_small(hipStream_t st, const bf16* A, const bf16* B, float* Cf, bf16* Cb,
                       const float* bias, const float* res, int M, int N, int K,
                       long lda, long ldb, long ldc, long sA, long sB, long sC, int batch)
{
    dim3 g((unsigned)(N / 64), (unsigned)(M / 64), (unsigned)batch);
    gemm_bf16_kernel<64,64,32,4,2,TA,TB,EPI><<<g, dim3(256), 0, st>>>(
        A, B, Cf, Cb, bias, res, M, N, K, lda, ldb, ldc, sA, sB, sC);
}

static inline unsigned nblk(long n) { return (unsigned)((n + 255) / 256); }

extern "C" void kernel_launch(void* const* d_in, const int* in_sizes, int n_in,
                              void* d_out, int out_size, void* d_ws, size_t ws_size,
                              hipStream_t stream)
{
    (void)in_sizes; (void)n_in; (void)out_size; (void)ws_size;

    const int  B = 32, T = 1024, F = 768, FFN = 3072, F2 = 1536, R = 64, MH = 3072;
    const long BT = (long)B * T;                 // 32768 rows

    const float* x      = (const float*)d_in[0];
    const float* la1_g  = (const float*)d_in[1];
    const float* la1_b  = (const float*)d_in[2];
    const float* U_w    = (const float*)d_in[3];
    const float* U_b    = (const float*)d_in[4];
    const float* norm_g = (const float*)d_in[5];
    const float* norm_b = (const float*)d_in[6];
    const float* bases0 = (const float*)d_in[7];
    const float* V_w    = (const float*)d_in[8];
    const float* V_b    = (const float*)d_in[9];
    const float* la2_g  = (const float*)d_in[10];
    const float* la2_b  = (const float*)d_in[11];
    const float* w1     = (const float*)d_in[12];
    const float* b1     = (const float*)d_in[13];
    const float* w2     = (const float*)d_in[14];
    const float* b2     = (const float*)d_in[15];
    float* out = (float*)d_out;

    // ---- bump allocator over d_ws ----
    char*  ws  = (char*)d_ws;
    size_t off = 0;
    auto carve = [&](size_t bytes) -> char* {
        char* p = ws + off;
        off += (bytes + 255) & ~(size_t)255;
        return p;
    };

    bf16*  h_bf    = (bf16*)carve(BT * F * 2);          // LN1 out; reused for LN2 (h2)
    float* u       = (float*)carve(BT * (long)FFN * 4); // gelu(h@U^T+b); reused for m1_bf
    bf16*  m1_bf   = (bf16*)u;                          // alias (u dead by then)
    bf16*  zs_bf   = (bf16*)carve(BT * (long)F2 * 2);   // relu(LN(z2)); reused for gate
    bf16*  g_bf    = zs_bf;                             // alias (zs dead by then)
    bf16*  wU_bf   = (bf16*)carve((long)FFN * F * 2);
    bf16*  wV_bf   = (bf16*)carve((long)F * F2 * 2);
    bf16*  w1_bf   = (bf16*)carve((long)MH * F * 2);
    bf16*  w2_bf   = (bf16*)carve((long)F * MH * 2);
    float* bases_f = (float*)carve((long)B * T * R * 4);
    bf16*  bases_b = (bf16*)carve((long)B * T * R * 2);
    float* coef_f  = (float*)carve((long)B * F2 * R * 4);
    bf16*  coef_b  = (bf16*)carve((long)B * F2 * R * 2);
    float* numc    = (float*)carve((long)B * F2 * R * 4); // also softmax score
    float* denc    = (float*)carve((long)B * F2 * R * 4);
    bf16*  btb_b   = (bf16*)carve((long)B * R * R * 2);   // basesT@bases / coefT@coef
    float* numb    = (float*)carve((long)B * T * R * 4);
    float* denb    = (float*)carve((long)B * T * R * 4);
    bf16*  z2r_b   = (bf16*)carve(BT * (long)F2 * 2);     // NMF reconstruction
    float* outf    = (float*)carve(BT * (long)F * 4);     // attn + x residual

    // ---- weight conversion f32 -> bf16 ----
    cvt_bf16_kernel<<<nblk((long)FFN * F), 256, 0, stream>>>(U_w, wU_bf, (long)FFN * F);
    cvt_bf16_kernel<<<nblk((long)F * F2), 256, 0, stream>>>(V_w, wV_bf, (long)F * F2);
    cvt_bf16_kernel<<<nblk((long)MH * F), 256, 0, stream>>>(w1, w1_bf, (long)MH * F);
    cvt_bf16_kernel<<<nblk((long)F * MH), 256, 0, stream>>>(w2, w2_bf, (long)F * MH);

    // ---- h = LN(x) ----
    ln_bf16_kernel<<<(unsigned)BT, 256, 0, stream>>>(x, F, la1_g, la1_b, h_bf, F, F, 0);

    // ---- u = gelu(h @ U_w^T + U_b)   (U_w is NxK row-major -> TB) ----
    gemm_big<false, true, 2>(stream, h_bf, wU_bf, u, nullptr, U_b, nullptr,
                             (int)BT, FFN, F, F, F, FFN, 0, 0, 0, 1);

    // ---- zs = relu(LN(z2)),  z2 = u[:, F2:] ----
    ln_bf16_kernel<<<(unsigned)BT, 256, 0, stream>>>(u + F2, FFN, norm_g, norm_b, zs_bf, F2, F2, 1);

    // ---- bases = broadcast(bases0) ----
    bases_init_kernel<<<nblk((long)B * T * R), 256, 0, stream>>>(
        bases0, bases_f, bases_b, (long)T * R, (long)B * T * R);

    const long sZS = (long)T * F2;   // zs per-batch stride
    const long sBA = (long)T * R;    // bases
    const long sCO = (long)F2 * R;   // coef / numc / denc / score
    const long sBT = (long)R * R;    // btb / ctc

    // ---- coef = softmax(INV_T * zs^T @ bases) ----
    gemm_small<true, false, 0>(stream, zs_bf, bases_b, numc, nullptr, nullptr, nullptr,
                               F2, R, T, F2, R, R, sZS, sBA, sCO, B);
    softmax64_kernel<<<(unsigned)((long)B * F2 / 8), 256, 0, stream>>>(
        numc, 100.0f, coef_f, coef_b, (long)B * F2);

    // ---- 6 multiplicative-update steps ----
    for (int it = 0; it < 6; ++it) {
        // num_c = zs^T @ bases        (B, F2, R)
        gemm_small<true, false, 0>(stream, zs_bf, bases_b, numc, nullptr, nullptr, nullptr,
                                   F2, R, T, F2, R, R, sZS, sBA, sCO, B);
        // btb = bases^T @ bases       (B, R, R) -> bf16
        gemm_small<true, false, 1>(stream, bases_b, bases_b, nullptr, btb_b, nullptr, nullptr,
                                   R, R, T, R, R, R, sBA, sBA, sBT, B);
        // den_c = coef @ btb          (B, F2, R)
        gemm_small<false, false, 0>(stream, coef_b, btb_b, denc, nullptr, nullptr, nullptr,
                                    F2, R, R, R, R, R, sCO, sBT, sCO, B);
        mul_update_kernel<<<nblk((long)B * F2 * R), 256, 0, stream>>>(
            coef_f, coef_b, numc, denc, (long)B * F2 * R);
        // num_b = zs @ coef           (B, T, R)
        gemm_small<false, false, 0>(stream, zs_bf, coef_b, numb, nullptr, nullptr, nullptr,
                                    T, R, F2, F2, R, R, sZS, sCO, sBA, B);
        // ctc = coef^T @ coef         (B, R, R) -> bf16 (reuse btb buffer)
        gemm_small<true, false, 1>(stream, coef_b, coef_b, nullptr, btb_b, nullptr, nullptr,
                                   R, R, F2, R, R, R, sCO, sCO, sBT, B);
        // den_b = bases @ ctc         (B, T, R)
        gemm_small<false, false, 0>(stream, bases_b, btb_b, denb, nullptr, nullptr, nullptr,
                                    T, R, R, R, R, R, sBA, sBT, sBA, B);
        mul_update_kernel<<<nblk((long)B * T * R), 256, 0, stream>>>(
            bases_f, bases_b, numb, denb, (long)B * T * R);
    }

    // ---- compute_coef: one more coef update (z == zs numerically) ----
    gemm_small<true, false, 0>(stream, zs_bf, bases_b, numc, nullptr, nullptr, nullptr,
                               F2, R, T, F2, R, R, sZS, sBA, sCO, B);
    gemm_small<true, false, 1>(stream, bases_b, bases_b, nullptr, btb_b, nullptr, nullptr,
                               R, R, T, R, R, R, sBA, sBA, sBT, B);
    gemm_small<false, false, 0>(stream, coef_b, btb_b, denc, nullptr, nullptr, nullptr,
                                F2, R, R, R, R, R, sCO, sBT, sCO, B);
    mul_update_kernel<<<nblk((long)B * F2 * R), 256, 0, stream>>>(
        coef_f, coef_b, numc, denc, (long)B * F2 * R);

    // ---- reconstruction: z2r = bases @ coef^T   (B, T, F2), bf16 ----
    gemm_big<false, true, 1>(stream, bases_b, coef_b, nullptr, z2r_b, nullptr, nullptr,
                             T, F2, R, R, R, F2, sBA, sCO, sZS, B);

    // ---- gate: g = z1 * z2r   (z1 = u[:, :F2]) ----
    gate_mul_kernel<<<nblk(BT * (long)F2), 256, 0, stream>>>(
        u, FFN, z2r_b, g_bf, BT * (long)F2, F2);

    // ---- out = g @ V_w^T + V_b + x ----
    gemm_big<false, true, 4>(stream, g_bf, wV_bf, outf, nullptr, V_b, x,
                             (int)BT, F, F2, F2, F2, F, 0, 0, 0, 1);

    // ---- h2 = LN(out) ----
    ln_bf16_kernel<<<(unsigned)BT, 256, 0, stream>>>(outf, F, la2_g, la2_b, h_bf, F, F, 0);

    // ---- m1 = gelu(h2 @ w1^T + b1), bf16 ----
    gemm_big<false, true, 3>(stream, h_bf, w1_bf, nullptr, m1_bf, b1, nullptr,
                             (int)BT, MH, F, F, F, MH, 0, 0, 0, 1);

    // ---- d_out = gelu(m1 @ w2^T + b2) + out ----
    gemm_big<false, true, 5>(stream, m1_bf, w2_bf, out, nullptr, b2, outf,
                             (int)BT, F, MH, MH, MH, F, 0, 0, 0, 1);
}
